// VMRNNCell_16990890623486
// MI455X (gfx1250) — compile-verified, matrix operand order checked by reading, not run on hardware
//
#include <hip/hip_runtime.h>
#include <hip/hip_bf16.h>
#include <cstdint>

// ---------------------------------------------------------------- constants
#define BATCH 16
#define HH    32
#define WW    32
#define LL    1024        // H*W
#define CC    256
#define DIN   512
#define NN    16
#define RR    16
#define BLC   ((size_t)BATCH * LL * CC)   // 4,194,304
#define EPSF  1e-5f

typedef __attribute__((ext_vector_type(16))) _Float16 v16h;
typedef __attribute__((ext_vector_type(8)))  _Float16 v8h;
typedef __attribute__((ext_vector_type(8)))  float    v8f;
typedef __attribute__((ext_vector_type(4)))  float    f4v;

// direction index maps (xs order -> natural l = h*W+w), H=W=32, L=1024
__device__ __forceinline__ int dirmap(int k, int l) {
    switch (k) {
        case 0: return l;
        case 1: return ((l & 31) << 5) | (l >> 5);
        case 2: return 1023 - l;
        default: { int j = 1023 - l; return ((j & 31) << 5) | (j >> 5); }
    }
}

__device__ __forceinline__ float sigmoidf_(float x) { return 1.f / (1.f + __expf(-x)); }
__device__ __forceinline__ float siluf_(float x)    { return x * sigmoidf_(x); }
__device__ __forceinline__ float softplusf_(float x){ return (x > 20.f) ? x : log1pf(__expf(x)); }

template<int NT>
__device__ __forceinline__ float block_sum(float v, float* red) {
    int t = threadIdx.x;
    red[t] = v; __syncthreads();
    #pragma unroll
    for (int s = NT / 2; s > 0; s >>= 1) {
        if (t < s) red[t] += red[t + s];
        __syncthreads();
    }
    float r = red[0]; __syncthreads();
    return r;
}

__device__ __forceinline__ v8h cvt8(const f4v& a, const f4v& b) {
    v8h h;
    h[0] = (_Float16)a[0]; h[1] = (_Float16)a[1];
    h[2] = (_Float16)a[2]; h[3] = (_Float16)a[3];
    h[4] = (_Float16)b[0]; h[5] = (_Float16)b[1];
    h[6] = (_Float16)b[2]; h[7] = (_Float16)b[3];
    return h;
}

// ---------------------------------------------------------------- WMMA GEMM
// Out[M,N] = A[M,K] * W[N,K]^T (+ bias)(+ addsrc), M = 16384 fixed.
// AMODE 0: A row-major (lda = K). AMODE 1: gather rows from xf (B,L,DIN) via dir map.
// Block tile 64x64, 8 wave32s, each wave owns 2 16x16 WMMA tiles, BK = 32,
// f32->f16 conversion fused into the LDS tile fill, LDS double-buffered.
//
// Fragment-layout key: for k = khi*16 + hs*8 + kk2 (khi,hs in {0,1}, kk2 in 0..7)
// the element lands at lane = (row&15) + hs*16, halfword e = khi*8 + kk2.
// => 8 consecutive (8-aligned) k of one row = one contiguous 16B LDS store.
template<int AMODE>
__global__ __launch_bounds__(256) void gemm_wmma(
    const float* __restrict__ A, const float* __restrict__ W,
    float* __restrict__ Out, int K, int N, int ldo,
    const float* __restrict__ bias, const float* __restrict__ addsrc, int dirk)
{
    __shared__ __align__(32) _Float16 Af[2][4][32][16];   // [buf][m-tile][lane][16 halves]
    __shared__ __align__(32) _Float16 Bf[2][4][32][16];   // [buf][n-tile][lane][16 halves]

    const int tid  = threadIdx.x;
    const int lane = tid & 31;
    const int wv   = tid >> 5;
    const int mt   = wv & 3;             // wave's m-tile 0..3
    const int nt0  = (wv >> 2) * 2;      // wave's first n-tile (0 or 2)
    const int row0 = blockIdx.y * 64;
    const int col0 = blockIdx.x * 64;

    // fill-stage decomposition: thread -> (row m, 8-wide k-group kg)
    const int m     = tid >> 2;          // 0..63
    const int kg    = tid & 3;           // 0..3  -> k offset kg*8
    const int khi   = kg >> 1;
    const int hsl   = kg & 1;
    const int ftile = m >> 4;
    const int flane = (m & 15) + (hsl << 4);
    const int fe    = khi * 8;           // halfword base within lane

    // source row pointers (k-invariant part)
    const float* arow;
    if (AMODE == 0) {
        arow = A + (size_t)(row0 + m) * K;
    } else {
        int r = row0 + m;
        int b = r >> 10, l = r & 1023;
        arow = A + ((size_t)((b << 10) | dirmap(dirk, l))) * DIN;
    }
    const bool nok  = (col0 + m) < N;
    const float* wrow = W + (size_t)(nok ? (col0 + m) : 0) * K;

    v8f acc0 = {}; v8f acc1 = {};
    f4v a0, a1, b0r, b1r;

    // prologue: load + stage tile kt = 0
    a0  = *(const f4v*)(arow + kg * 8);
    a1  = *(const f4v*)(arow + kg * 8 + 4);
    b0r = nok ? *(const f4v*)(wrow + kg * 8)     : (f4v){};
    b1r = nok ? *(const f4v*)(wrow + kg * 8 + 4) : (f4v){};
    *(v8h*)(&Af[0][ftile][flane][fe]) = cvt8(a0, a1);
    *(v8h*)(&Bf[0][ftile][flane][fe]) = cvt8(b0r, b1r);

    int cur = 0;
    for (int kt = 32;; kt += 32) {
        __syncthreads();
        const bool more = kt < K;
        if (more) {                       // issue next tile's global loads early
            a0  = *(const f4v*)(arow + kt + kg * 8);
            a1  = *(const f4v*)(arow + kt + kg * 8 + 4);
            b0r = nok ? *(const f4v*)(wrow + kt + kg * 8)     : (f4v){};
            b1r = nok ? *(const f4v*)(wrow + kt + kg * 8 + 4) : (f4v){};
        }
        // matrix math on current LDS buffer (hides the global-load latency)
        v16h a  = *(const v16h*)(&Af[cur][mt][lane][0]);
        v16h bb0 = *(const v16h*)(&Bf[cur][nt0][lane][0]);
        v16h bb1 = *(const v16h*)(&Bf[cur][nt0 + 1][lane][0]);
        acc0 = __builtin_amdgcn_wmma_f32_16x16x32_f16(false, a, false, bb0,
                                                      (short)0, acc0, false, false);
        acc1 = __builtin_amdgcn_wmma_f32_16x16x32_f16(false, a, false, bb1,
                                                      (short)0, acc1, false, false);
        if (!more) break;
        // convert + stage into the other buffer
        *(v8h*)(&Af[cur ^ 1][ftile][flane][fe]) = cvt8(a0, a1);
        *(v8h*)(&Bf[cur ^ 1][ftile][flane][fe]) = cvt8(b0r, b1r);
        cur ^= 1;
    }

    // epilogue: v8f layout -> lanes 0-15: M=r, N=lane; lanes 16-31: M=8+r, N=lane-16
    const int hs = lane >> 4, nloc = lane & 15;
    #pragma unroll
    for (int r = 0; r < 8; ++r) {
        int gm = row0 + mt * 16 + r + 8 * hs;
        #pragma unroll
        for (int t = 0; t < 2; ++t) {
            int gn = col0 + (nt0 + t) * 16 + nloc;
            if (gn < N) {
                float v = (t == 0) ? acc0[r] : acc1[r];
                if (bias)   v += bias[gn];
                if (addsrc) v += addsrc[(size_t)gm * ldo + gn];
                Out[(size_t)gm * ldo + gn] = v;
            }
        }
    }
}

// ---------------------------------------------------------------- LN kernels
// layer0: cat = [LN(x), LN(hx)]  (per token, C=256 each)
__global__ __launch_bounds__(256) void ln_cat_kernel(
    const float* __restrict__ x, const float* __restrict__ hx,
    const float* __restrict__ g, const float* __restrict__ bb,
    float* __restrict__ cat)
{
    __shared__ float red[256];
    const int bl = blockIdx.x, c = threadIdx.x;
    #pragma unroll
    for (int s = 0; s < 2; ++s) {
        const float* src = (s == 0) ? x : hx;
        float xv  = src[(size_t)bl * CC + c];
        float mu  = block_sum<256>(xv, red) * (1.f / CC);
        float d0  = xv - mu;
        float var = block_sum<256>(d0 * d0, red) * (1.f / CC);
        float xn  = d0 * rsqrtf(var + EPSF) * g[c] + bb[c];
        cat[(size_t)bl * (2 * CC) + s * CC + c] = xn;
    }
}

// layer1: xn = LN(x)
__global__ __launch_bounds__(256) void ln_kernel(
    const float* __restrict__ x, const float* __restrict__ g,
    const float* __restrict__ bb, float* __restrict__ xn)
{
    __shared__ float red[256];
    const int bl = blockIdx.x, c = threadIdx.x;
    float xv  = x[(size_t)bl * CC + c];
    float mu  = block_sum<256>(xv, red) * (1.f / CC);
    float d0  = xv - mu;
    float var = block_sum<256>(d0 * d0, red) * (1.f / CC);
    xn[(size_t)bl * CC + c] = d0 * rsqrtf(var + EPSF) * g[c] + bb[c];
}

// --------------------------------------------------- depthwise conv3x3+SiLU
// xz (B,L,2*DIN) cols 0..511 -> xf (B,L,DIN), one block per token, lane = d
__global__ __launch_bounds__(512) void conv_silu_kernel(
    const float* __restrict__ xz, const float* __restrict__ cw,
    const float* __restrict__ cb, float* __restrict__ xf)
{
    const int d  = threadIdx.x;
    const int bl = blockIdx.x;
    const int l = bl & 1023, b = bl >> 10;
    const int h = l >> 5, w = l & 31;
    float s = cb[d];
    #pragma unroll
    for (int kh = 0; kh < 3; ++kh) {
        int hh = h + kh - 1; if (hh < 0 || hh > 31) continue;
        #pragma unroll
        for (int kw = 0; kw < 3; ++kw) {
            int ww2 = w + kw - 1; if (ww2 < 0 || ww2 > 31) continue;
            int ls = (hh << 5) | ww2;
            s += xz[((size_t)((b << 10) | ls)) * (2 * DIN) + d] * cw[d * 9 + kh * 3 + kw];
        }
    }
    xf[(size_t)bl * DIN + d] = siluf_(s);
}

// ------------------------------------------------------------ selective scan
// one block per (b, direction k); lane d owns channel (k*512+d), h[16] in regs.
// x_dbl rows (48-wide: [dts(16) | B(16) | C(16)]) staged into LDS per 64 steps.
__global__ __launch_bounds__(512) void scan_kernel(
    const float* __restrict__ xdbl, const float* __restrict__ xf,
    const float* __restrict__ Alogs, const float* __restrict__ Dsv,
    const float* __restrict__ dtw_g, const float* __restrict__ dtb_g,
    float* __restrict__ ybuf)
{
    __shared__ float rows[64 * 48];     // 12 KB
    const int d = threadIdx.x;
    const int b = blockIdx.x >> 2, k = blockIdx.x & 3;
    const int ch = k * DIN + d;

    float Ar[NN], dtw[RR], hst[NN];
    #pragma unroll
    for (int n = 0; n < NN; ++n) { Ar[n] = -__expf(Alogs[(size_t)ch * NN + n]); hst[n] = 0.f; }
    #pragma unroll
    for (int r = 0; r < RR; ++r) dtw[r] = dtw_g[(size_t)ch * RR + r];
    const float dtb = dtb_g[ch];
    const float Dd  = Dsv[ch];
    const float* base = xdbl + (size_t)(k * BATCH + b) * LL * 48;

    for (int lc = 0; lc < LL; lc += 64) {
        __syncthreads();
        for (int i = d; i < 64 * 48; i += 512) rows[i] = base[(size_t)lc * 48 + i];
        __syncthreads();
        for (int ll = 0; ll < 64; ++ll) {
            const float* row = &rows[ll * 48];
            const int l = lc + ll;
            float dt = dtb;
            #pragma unroll
            for (int r = 0; r < RR; ++r) dt += row[r] * dtw[r];
            const float delta = softplusf_(dt);
            const int   mp = dirmap(k, l);
            const float u  = xf[((size_t)((b << 10) | mp)) * DIN + d];
            const float du = delta * u;
            float y = 0.f;
            #pragma unroll
            for (int n = 0; n < NN; ++n) {
                hst[n] = __expf(delta * Ar[n]) * hst[n] + du * row[16 + n];
                y += hst[n] * row[32 + n];
            }
            atomicAdd(&ybuf[((size_t)((b << 10) | mp)) * DIN + d], y + Dd * u);
        }
    }
}

// ------------------------------------------- out-norm (over DIN) + SiLU gate
__global__ __launch_bounds__(256) void post_kernel(
    const float* __restrict__ y, const float* __restrict__ xz,
    const float* __restrict__ g, const float* __restrict__ bb,
    float* __restrict__ yg)
{
    __shared__ float red[256];
    const int bl = blockIdx.x, c = threadIdx.x;
    float y0 = y[(size_t)bl * DIN + c];
    float y1 = y[(size_t)bl * DIN + 256 + c];
    float mu = block_sum<256>(y0 + y1, red) * (1.f / DIN);
    float e0 = y0 - mu, e1 = y1 - mu;
    float var = block_sum<256>(e0 * e0 + e1 * e1, red) * (1.f / DIN);
    float rs  = rsqrtf(var + EPSF);
    #pragma unroll
    for (int s = 0; s < 2; ++s) {
        int   dd = c + s * 256;
        float ev = (s == 0) ? e0 : e1;
        float yn = ev * rs * g[dd] + bb[dd];
        float z  = xz[(size_t)bl * (2 * DIN) + DIN + dd];
        yg[(size_t)bl * DIN + dd] = yn * siluf_(z);
    }
}

// ------------------------------------------------------------- LSTM gating
__global__ __launch_bounds__(256) void final_gate_kernel(
    const float* __restrict__ o, const float* __restrict__ cx,
    float* __restrict__ Ht, float* __restrict__ Ct, size_t n)
{
    size_t i = (size_t)blockIdx.x * 256 + threadIdx.x;
    if (i >= n) return;
    float ov = o[i];
    float F  = sigmoidf_(ov);
    float c  = F * (cx[i] + tanhf(ov));
    Ht[i] = F * tanhf(c);
    Ct[i] = c;
}

// ---------------------------------------------------------------- launcher
extern "C" void kernel_launch(void* const* d_in, const int* in_sizes, int n_in,
                              void* d_out, int out_size, void* d_ws, size_t ws_size,
                              hipStream_t stream) {
    const float* xt       = (const float*)d_in[0];
    const float* hx       = (const float*)d_in[1];
    const float* cx       = (const float*)d_in[2];
    const float* ln1_g    = (const float*)d_in[3];
    const float* ln1_b    = (const float*)d_in[4];
    const float* lin_w    = (const float*)d_in[5];
    const float* lin_b    = (const float*)d_in[6];
    const float* inproj_w = (const float*)d_in[7];
    const float* conv_w   = (const float*)d_in[8];
    const float* conv_b   = (const float*)d_in[9];
    const float* xproj_w  = (const float*)d_in[10];
    const float* dtproj_w = (const float*)d_in[11];
    const float* dtproj_b = (const float*)d_in[12];
    const float* A_logs   = (const float*)d_in[13];
    const float* Ds       = (const float*)d_in[14];
    const float* outn_g   = (const float*)d_in[15];
    const float* outn_b   = (const float*)d_in[16];
    const float* outp_w   = (const float*)d_in[17];

    float* out = (float*)d_out;
    float* ws  = (float*)d_ws;

    const size_t M = (size_t)BATCH * LL;   // 16384 rows
    size_t off = 0;
    float* buf_cat  = ws + off; off += M * (2 * CC);   // also reused as yg
    float* buf_xn   = ws + off; off += M * CC;
    float* buf_xz   = ws + off; off += M * (2 * DIN);
    float* buf_xf   = ws + off; off += M * DIN;
    float* buf_xdbl = ws + off; off += (size_t)4 * M * 48;
    float* buf_y    = ws + off; off += M * DIN;
    float* buf_x    = ws + off; off += M * CC;
    float* buf_yg   = buf_cat;                         // alias (cat dead by then)

    const int gridBL = BATCH * LL;                     // 16384 token blocks
    const dim3 blk256(256), blk512(512);

    for (int lay = 0; lay < 2; ++lay) {
        const float* shortcut = (lay == 0) ? xt : buf_x;

        // ---- LayerNorm (+ hidden concat + lin on layer 0) -> buf_xn (B,L,C)
        if (lay == 0) {
            ln_cat_kernel<<<gridBL, blk256, 0, stream>>>(xt, hx, ln1_g, ln1_b, buf_cat);
            gemm_wmma<0><<<dim3(CC / 64, M / 64), blk256, 0, stream>>>(
                buf_cat, lin_w, buf_xn, 2 * CC, CC, CC, lin_b, nullptr, 0);
        } else {
            ln_kernel<<<gridBL, blk256, 0, stream>>>(buf_x, ln1_g + CC, ln1_b + CC, buf_xn);
        }

        // ---- inproj: (M,256) x (1024,256)^T -> xz (M,1024)
        gemm_wmma<0><<<dim3((2 * DIN) / 64, M / 64), blk256, 0, stream>>>(
            buf_xn, inproj_w + (size_t)lay * (2 * DIN) * CC, buf_xz,
            CC, 2 * DIN, 2 * DIN, nullptr, nullptr, 0);

        // ---- depthwise conv3x3 + SiLU -> xf (B,L,DIN)
        conv_silu_kernel<<<gridBL, blk512, 0, stream>>>(
            buf_xz, conv_w + (size_t)lay * DIN * 9, conv_b + (size_t)lay * DIN, buf_xf);

        // ---- xproj per direction: gathered (M,512) x (48,512)^T -> xdbl[k]
        for (int k = 0; k < 4; ++k) {
            gemm_wmma<1><<<dim3(1, M / 64), blk256, 0, stream>>>(
                buf_xf, xproj_w + (size_t)(lay * 4 + k) * 48 * DIN,
                buf_xdbl + (size_t)k * M * 48, DIN, 48, 48, nullptr, nullptr, k);
        }

        // ---- selective scan (4 directions fused via atomics into buf_y)
        hipMemsetAsync(buf_y, 0, M * DIN * sizeof(float), stream);
        scan_kernel<<<BATCH * 4, blk512, 0, stream>>>(
            buf_xdbl, buf_xf,
            A_logs   + (size_t)lay * 4 * DIN * NN,
            Ds       + (size_t)lay * 4 * DIN,
            dtproj_w + (size_t)lay * 4 * DIN * RR,
            dtproj_b + (size_t)lay * 4 * DIN,
            buf_y);

        // ---- out-norm + SiLU(z) gate -> yg (M,512)
        post_kernel<<<gridBL, blk256, 0, stream>>>(
            buf_y, buf_xz, outn_g + (size_t)lay * DIN, outn_b + (size_t)lay * DIN, buf_yg);

        // ---- outproj + residual: (M,512) x (256,512)^T + shortcut -> buf_x
        gemm_wmma<0><<<dim3(CC / 64, M / 64), blk256, 0, stream>>>(
            buf_yg, outp_w + (size_t)lay * CC * DIN, buf_x,
            DIN, CC, CC, nullptr, shortcut, 0);
    }

    // ---- LSTM-style gate: Ht, Ct
    final_gate_kernel<<<(unsigned)((BLC + 255) / 256), blk256, 0, stream>>>(
        buf_x, cx, out, out + BLC, BLC);
}
